// PAWSNonParametric_70531952935510
// MI455X (gfx1250) — matrix operand order repaired and checked by reading, not compiled, not run on hardware
//
#include <hip/hip_runtime.h>
#include <math.h>

// ---------------------------------------------------------------------------
// PAWS non-parametric pseudo-label, fused flash-softmax formulation, CDNA5.
//   branch b: out[u, c] = sum_{s : label_b[s]==c} softmax_s( <q_u, k_s> / tau )
// bf16 WMMA (16x16x32) for the 8192x8192x512 similarity GEMMs, f32 softmax.
// Both passes stream support tiles through LDS with async global->LDS copies
// (ASYNCcnt, double buffered), shared across all waves of a block.
// ---------------------------------------------------------------------------

typedef __attribute__((ext_vector_type(16))) __bf16        v16bf;
typedef __attribute__((ext_vector_type(8)))  float         v8f;
typedef __attribute__((ext_vector_type(4)))  unsigned int  u32x4;

union FragBF { v16bf v; u32x4 q[2]; };

#define TAU_INV 10.0f

static constexpr int NU = 8192;   // unlabeled rows (anchor / positive)
static constexpr int NS = 8192;   // support rows per branch (16384 / 2)
static constexpr int DD = 512;    // feature dim
static constexpr int CC = 1000;   // classes
static constexpr int RS = 520;    // padded LDS row stride (ushorts): 1040 B

static __device__ __forceinline__ unsigned short f32_to_bf16(float f) {
  unsigned int u = __float_as_uint(f);
  unsigned int r = 0x7FFFu + ((u >> 16) & 1u);   // round-to-nearest-even
  return (unsigned short)((u + r) >> 16);
}

// ---------------------------------------------------------------------------
// 1) Row L2-normalize f32 -> bf16.  One block (128 threads) per row.
// ---------------------------------------------------------------------------
__global__ void __launch_bounds__(128)
rownorm_bf16_kernel(const float* __restrict__ in, unsigned short* __restrict__ out) {
  const int row = blockIdx.x;
  const float* x = in + (size_t)row * DD;
  float ss = 0.f;
  for (int i = threadIdx.x; i < DD; i += 128) { float v = x[i]; ss += v * v; }
  __shared__ float red[4];
  for (int off = 16; off > 0; off >>= 1) ss += __shfl_xor(ss, off, 32);
  if ((threadIdx.x & 31) == 0) red[threadIdx.x >> 5] = ss;
  __syncthreads();
  const float tot = red[0] + red[1] + red[2] + red[3];
  const float inv = 1.0f / fmaxf(sqrtf(tot), 1e-12f);
  unsigned short* o = out + (size_t)row * DD;
  for (int i = threadIdx.x; i < DD; i += 128) o[i] = f32_to_bf16(x[i] * inv);
}

// ---------------------------------------------------------------------------
// 2) Recover labels from one-hot: argmax per row.  One wave32 per row.
// ---------------------------------------------------------------------------
__global__ void __launch_bounds__(128)
argmax_kernel(const float* __restrict__ oh, int* __restrict__ labels) {
  const int row  = blockIdx.x * 4 + (threadIdx.x >> 5);
  const int lane = threadIdx.x & 31;
  const float* r = oh + (size_t)row * CC;
  float best = -1.0f; int bi = 0;
  for (int c = lane; c < CC; c += 32) {
    float v = r[c];
    if (v > best) { best = v; bi = c; }
  }
  for (int off = 16; off > 0; off >>= 1) {
    float ov = __shfl_xor(best, off, 32);
    int   oi = __shfl_xor(bi,   off, 32);
    if (ov > best || (ov == best && oi < bi)) { best = ov; bi = oi; }
  }
  if (lane == 0) labels[row] = bi;
}

// ---------------------------------------------------------------------------
// WMMA fragment loaders (bf16, 16x16x32; layouts per CDNA5 ISA 7.12.2).
//   A 16x32: lane l -> row l%16, K = 8*(l/16) + {0..7} and +16..+23
//   B 32x16: lane l -> col l%16, K = 8*(l/16) + {0..7} and +16..+23
// ---------------------------------------------------------------------------
static __device__ __forceinline__ FragBF load_fragA(const unsigned short* rowPtr,
                                                    int kk, int half) {
  FragBF f;
  const unsigned short* p = rowPtr + kk + 8 * half;
  f.q[0] = *(const u32x4*)(p);
  f.q[1] = *(const u32x4*)(p + 16);
  return f;
}
static __device__ __forceinline__ FragBF load_fragB(const unsigned short* rowPtr,
                                                    int kk, int half) {
  FragBF f;
  const unsigned short* p = rowPtr + kk + 8 * half;
  f.q[0] = *(const u32x4*)(p);
  f.q[1] = *(const u32x4*)(p + 16);
  return f;
}

// ---------------------------------------------------------------------------
// Async global->LDS staging of one 16-row x 512-col bf16 support tile into a
// padded LDS buffer (16 KB), tracked on ASYNCcnt; async loads complete in
// order, so a double-buffer wait only needs "<= per-tile issue count".
// LDS operand = low 32 bits of the flat address (aperture low bits are the
// wave-relative LDS offset per the CDNA5 aperture rules).
// ---------------------------------------------------------------------------
template <int NTHREADS>
static __device__ __forceinline__ void stage_tile_async(
    const unsigned short* __restrict__ Km, int n0,
    unsigned short* dstBase, int tid) {
#pragma unroll
  for (int i = 0; i < 1024 / NTHREADS; ++i) {
    const int c   = i * NTHREADS + tid;   // 16B-chunk id, 0..1023
    const int row = c >> 6;               // 0..15
    const int col = (c & 63) * 8;         // ushort offset within row
    const unsigned long long ga =
        (unsigned long long)(size_t)(Km + (size_t)(n0 + row) * DD + col);
    const unsigned dst = (unsigned)(size_t)(dstBase + row * RS + col);
    asm volatile("global_load_async_to_lds_b128 %0, %1, off"
                 :: "v"(dst), "v"(ga) : "memory");
  }
}
#define WAIT_ASYNC(hex) asm volatile("s_wait_asynccnt " hex ::: "memory")

// ---------------------------------------------------------------------------
// 3) Pass A: streaming online-softmax stats (row max, row sumexp).
//    Block = 128 threads (4 waves, 64 query rows). A fragments preloaded to
//    registers (16 frags x 8 VGPRs); B tiles double-buffered in LDS via
//    async copies and shared by all 4 waves (4x less global B traffic).
// ---------------------------------------------------------------------------
__global__ void __launch_bounds__(128)
softmax_stats_kernel(const unsigned short* __restrict__ Q,
                     const unsigned short* __restrict__ Km,
                     float* __restrict__ rmax, float* __restrict__ rsum) {
  __shared__ __align__(16) unsigned short smemB[2][16 * RS];   // 2 x 16.25 KB
  const int m0   = blockIdx.x * 64;
  const int tid  = threadIdx.x;
  const int wave = tid >> 5;
  const int lane = tid & 31;
  const int half = lane >> 4;

  // Preload this wave's 16-row A tile fragments into registers.
  FragBF afr[16];
  {
    const unsigned short* Arow =
        Q + (size_t)(m0 + wave * 16 + (lane & 15)) * DD;
#pragma unroll
    for (int k16 = 0; k16 < 16; ++k16) afr[k16] = load_fragA(Arow, k16 * 32, half);
  }

  float runm[8], runs[8];
#pragma unroll
  for (int j = 0; j < 8; ++j) { runm[j] = -INFINITY; runs[j] = 0.f; }

  constexpr int NT = NS / 16;                      // 512 support tiles
  stage_tile_async<128>(Km, 0, smemB[0], tid);     // prologue

  for (int t = 0; t < NT; ++t) {
    const int cur = t & 1;
    if (t + 1 < NT) {
      stage_tile_async<128>(Km, (t + 1) * 16, smemB[cur ^ 1], tid);
      WAIT_ASYNC("0x8");                           // tile t complete (in-order)
    } else {
      WAIT_ASYNC("0x0");
    }
    __syncthreads();                               // LDS visible to all waves

    const unsigned short* Brow = smemB[cur] + (lane & 15) * RS;
    v8f acc = {};
#pragma unroll
    for (int kk = 0; kk < DD; kk += 32) {
      FragBF b = load_fragB(Brow, kk, half);       // ds_load_b128 x2
      acc = __builtin_amdgcn_wmma_f32_16x16x32_bf16(
          false, afr[kk / 32].v, false, b.v, (short)0, acc, false, false);
    }
#pragma unroll
    for (int j = 0; j < 8; ++j) {                  // online softmax update
      float x  = acc[j] * TAU_INV;
      float nm = fmaxf(runm[j], x);
      runs[j]  = runs[j] * __expf(runm[j] - nm) + __expf(x - nm);
      runm[j]  = nm;
    }
    __syncthreads();                               // done reading cur buffer
  }

  // Combine the 16 lanes sharing each output row (xor stays within halves).
#pragma unroll
  for (int j = 0; j < 8; ++j) {
    float m = runm[j], s = runs[j];
#pragma unroll
    for (int off = 1; off < 16; off <<= 1) {
      float om = __shfl_xor(m, off, 32);
      float os = __shfl_xor(s, off, 32);
      float nm = fmaxf(m, om);
      s = s * __expf(m - nm) + os * __expf(om - nm);
      m = nm;
    }
    if ((lane & 15) == 0) {
      const int row = m0 + wave * 16 + j + 8 * half;
      rmax[row] = m;
      rsum[row] = s;
    }
  }
}

// ---------------------------------------------------------------------------
// 4) Pass B: recompute tiles, p = exp(s/tau - m)/Z, scatter-add by label into
//    per-row f32 LDS class bins (ds_add_f32), then one coalesced store.
//    Block = 64 threads (2 waves, 32 query rows). Both waves consume the SAME
//    async-staged B tile (distinct A row-tiles -> no duplicated WMMA), double
//    buffered exactly like pass A.  LDS: 125 KB bins + 32.5 KB B buffers.
// ---------------------------------------------------------------------------
__global__ void __launch_bounds__(64)
scatter_kernel(const unsigned short* __restrict__ Q,
               const unsigned short* __restrict__ Km,
               const int* __restrict__ labels,
               const float* __restrict__ rmax, const float* __restrict__ rsum,
               float* __restrict__ out) {
  __shared__ float smemOut[32 * CC];                           // 125 KB
  __shared__ __align__(16) unsigned short smemB[2][16 * RS];   // 2 x 16.25 KB
  const int m0   = blockIdx.x * 32;
  const int tid  = threadIdx.x;
  const int wave = tid >> 5;
  const int lane = tid & 31;
  const int half = lane >> 4;

  for (int i = tid; i < 32 * CC; i += 64) smemOut[i] = 0.f;

  float rm[8], rinv[8];
#pragma unroll
  for (int j = 0; j < 8; ++j) {
    const int row = m0 + wave * 16 + j + 8 * half;
    rm[j]   = rmax[row];
    rinv[j] = 1.0f / rsum[row];        // >= 1, never zero
  }

  FragBF afr[16];
  {
    const unsigned short* Arow =
        Q + (size_t)(m0 + wave * 16 + (lane & 15)) * DD;
#pragma unroll
    for (int k16 = 0; k16 < 16; ++k16) afr[k16] = load_fragA(Arow, k16 * 32, half);
  }

  constexpr int NT = NS / 16;                      // 512 support tiles
  stage_tile_async<64>(Km, 0, smemB[0], tid);      // prologue (16 issues/wave)
  __syncthreads();                                 // smemOut zero-init visible

  for (int t = 0; t < NT; ++t) {
    const int cur = t & 1;
    if (t + 1 < NT) {
      stage_tile_async<64>(Km, (t + 1) * 16, smemB[cur ^ 1], tid);
      WAIT_ASYNC("0x10");                          // tile t complete (in-order)
    } else {
      WAIT_ASYNC("0x0");
    }
    __syncthreads();                               // LDS visible to all waves

    const unsigned short* Brow = smemB[cur] + (lane & 15) * RS;
    v8f acc = {};
#pragma unroll
    for (int kk = 0; kk < DD; kk += 32) {
      FragBF b = load_fragB(Brow, kk, half);
      acc = __builtin_amdgcn_wmma_f32_16x16x32_bf16(
          false, afr[kk / 32].v, false, b.v, (short)0, acc, false, false);
    }
    const int cls = labels[t * 16 + (lane & 15)];
#pragma unroll
    for (int j = 0; j < 8; ++j) {
      const float p = __expf(acc[j] * TAU_INV - rm[j]) * rinv[j];
      atomicAdd(&smemOut[(wave * 16 + j + 8 * half) * CC + cls], p);  // ds_add_f32
    }
    __syncthreads();                               // done reading cur buffer
  }
  __syncthreads();

  float* outBase = out + (size_t)m0 * CC;
  for (int i = tid; i < 32 * CC; i += 64) outBase[i] = smemOut[i];
}

// ---------------------------------------------------------------------------
// Host launcher
// ---------------------------------------------------------------------------
extern "C" void kernel_launch(void* const* d_in, const int* in_sizes, int n_in,
                              void* d_out, int out_size, void* d_ws, size_t ws_size,
                              hipStream_t stream) {
  const float* anchor   = (const float*)d_in[0];   // NU x DD
  const float* positive = (const float*)d_in[1];   // NU x DD
  const float* lb_feat  = (const float*)d_in[2];   // 2*NS x DD
  const float* lb_oh    = (const float*)d_in[3];   // 2*NS x CC
  const float* logits   = (const float*)d_in[4];   // 2*NS x CC
  float* out = (float*)d_out;

  // Output offsets (floats), in reference return order.
  const size_t o1 = (size_t)NU * DD;            // positive_feat
  const size_t o2 = o1 + (size_t)NU * DD;       // lb_feat
  const size_t o3 = o2 + (size_t)2 * NS * DD;   // lb_one_hot
  const size_t o4 = o3 + (size_t)2 * NS * CC;   // logits_x_lb
  const size_t o5 = o4 + (size_t)2 * NS * CC;   // out_ulb_1
  const size_t o6 = o5 + (size_t)NU * CC;       // out_ulb_2

  // Workspace carve-up (all 16B-aligned chunks).
  char* ws = (char*)d_ws;
  unsigned short* qa = (unsigned short*)ws;  ws += (size_t)NU * DD * 2;
  unsigned short* qp = (unsigned short*)ws;  ws += (size_t)NU * DD * 2;
  unsigned short* k1 = (unsigned short*)ws;  ws += (size_t)NS * DD * 2;
  unsigned short* k2 = (unsigned short*)ws;  ws += (size_t)NS * DD * 2;
  int* lab1 = (int*)ws;                      ws += (size_t)NS * 4;
  int* lab2 = (int*)ws;                      ws += (size_t)NS * 4;
  float* rmax1 = (float*)ws;                 ws += (size_t)NU * 4;
  float* rsum1 = (float*)ws;                 ws += (size_t)NU * 4;
  float* rmax2 = (float*)ws;                 ws += (size_t)NU * 4;
  float* rsum2 = (float*)ws;                 ws += (size_t)NU * 4;

  // Pass-through outputs (async D2D copies are graph-capture safe).
  hipMemcpyAsync(out,      anchor,   (size_t)NU * DD * 4,     hipMemcpyDeviceToDevice, stream);
  hipMemcpyAsync(out + o1, positive, (size_t)NU * DD * 4,     hipMemcpyDeviceToDevice, stream);
  hipMemcpyAsync(out + o2, lb_feat,  (size_t)2 * NS * DD * 4, hipMemcpyDeviceToDevice, stream);
  hipMemcpyAsync(out + o3, lb_oh,    (size_t)2 * NS * CC * 4, hipMemcpyDeviceToDevice, stream);
  hipMemcpyAsync(out + o4, logits,   (size_t)2 * NS * CC * 4, hipMemcpyDeviceToDevice, stream);

  // Normalize + bf16 convert (queries and both support halves).
  rownorm_bf16_kernel<<<NU, 128, 0, stream>>>(anchor,   qa);
  rownorm_bf16_kernel<<<NU, 128, 0, stream>>>(positive, qp);
  rownorm_bf16_kernel<<<NS, 128, 0, stream>>>(lb_feat,                   k1);
  rownorm_bf16_kernel<<<NS, 128, 0, stream>>>(lb_feat + (size_t)NS * DD, k2);

  // Labels from one-hot halves.
  argmax_kernel<<<NS / 4, 128, 0, stream>>>(lb_oh,                   lab1);
  argmax_kernel<<<NS / 4, 128, 0, stream>>>(lb_oh + (size_t)NS * CC, lab2);

  // Pass A: softmax stats (WMMA bf16 GEMM, async double-buffered B tiles).
  softmax_stats_kernel<<<NU / 64, 128, 0, stream>>>(qa, k1, rmax1, rsum1);
  softmax_stats_kernel<<<NU / 64, 128, 0, stream>>>(qp, k2, rmax2, rsum2);

  // Pass B: probability scatter into class bins (async-staged shared B).
  scatter_kernel<<<NU / 32, 64, 0, stream>>>(qa, k1, lab1, rmax1, rsum1, out + o5);
  scatter_kernel<<<NU / 32, 64, 0, stream>>>(qp, k2, lab2, rmax2, rsum2, out + o6);

  (void)in_sizes; (void)n_in; (void)out_size; (void)ws_size; (void)o6;
}
